// SipmResponse_14791867367484
// MI455X (gfx1250) — matrix operand order repaired and checked by reading, not compiled, not run on hardware
//
#include <hip/hip_runtime.h>
#include <hip/hip_fp16.h>

typedef __attribute__((ext_vector_type(16))) _Float16 v16h;
typedef __attribute__((ext_vector_type(8)))  _Float16 v8h;
typedef __attribute__((ext_vector_type(8)))  float    v8f;

#define B_   4
#define N_   1024
#define HW_  48
#define T_   1024
#define HID_ 16
#define K_   (B_ * N_)     /* 4096 = 2^12 */
#define S_   (HW_ * HW_)   /* 2304 */

// ---------------------------------------------------------------------------
// Phase 1: per (electron k, sensor s) PSF response, written TRANSPOSED as
// RT[s*K + k] in f16 so the GEMM B-fragment loads are contiguous along K.
// ---------------------------------------------------------------------------
__global__ __launch_bounds__(256) void psf_kernel(
    const float* __restrict__ el_photons,   // (B,N,1)
    const float* __restrict__ xy,           // (B,N,2)
    const float* __restrict__ sensors,      // (H,W,2)
    const float* __restrict__ W1, const float* __restrict__ b1,
    const float* __restrict__ W2, const float* __restrict__ b2,
    const float* __restrict__ W3, const float* __restrict__ b3,
    const float* __restrict__ amplitude,
    _Float16* __restrict__ RT)
{
    const int tid = blockIdx.x * 256 + threadIdx.x;
    const int k = tid & (K_ - 1);      // electron index (b*N+n), fastest
    const int s = tid >> 12;           // sensor index
    if (s >= S_) return;

    const float px = xy[2 * k + 0], py = xy[2 * k + 1];
    const float sx = sensors[2 * s + 0], sy = sensors[2 * s + 1];
    const float dx = px - sx, dy = py - sy;
    const float r2 = dx * dx + dy * dy;
    const float baseline = __expf(-0.01f * r2);
    const float r = __fsqrt_rn(r2) * (1.0f / 500.0f);

    // scalar-input MLP: 1 -> 16 -> 16 -> 1 (weights uniform -> scalar loads)
    float h[HID_];
#pragma unroll
    for (int j = 0; j < HID_; ++j) h[j] = tanhf(fmaf(r, W1[j], b1[j]));

    float psf = b3[0];
#pragma unroll
    for (int i = 0; i < HID_; ++i) {
        float acc = b2[i];
#pragma unroll
        for (int j = 0; j < HID_; ++j) acc = fmaf(h[j], W2[j * HID_ + i], acc);
        psf = fmaf(tanhf(acc), W3[i], psf);
    }

    const float amp = __expf(amplitude[0]);
    const float val = el_photons[k] * amp * baseline * (1.0f + psf);
    RT[(size_t)s * K_ + k] = (_Float16)val;
}

// ---------------------------------------------------------------------------
// Phase 2: gaussian time weights, written TRANSPOSED as GT[t*K + k] in f16 so
// the GEMM A-fragment loads are contiguous along K.
// ---------------------------------------------------------------------------
__global__ __launch_bounds__(256) void gauss_kernel(
    const float* __restrict__ z,        // (B,N)
    _Float16* __restrict__ GT)
{
    const int tid = blockIdx.x * 256 + threadIdx.x; // t*K + k
    const int k = tid & (K_ - 1);
    const int t = tid >> 12;
    const float d = (float)t - z[k];
    const float g = __expf(-0.5f * d * d) * 0.3989422804f; // sigma = 1
    GT[tid] = (_Float16)g;
}

// ---------------------------------------------------------------------------
// Phase 3: OUT[t,s] = sum_k GT[t,k] * RT[s,k]  via V_WMMA_F32_16X16X32_F16.
// Each wave computes a 32(M=t) x 64(N=s) macro-tile: 2x4 accumulators,
// 8 WMMAs per K-step of 32. Output stored as d_out[s*T + t]  ((H,W,T) order).
// ---------------------------------------------------------------------------
#define MT 2
#define NT 4
__global__ __launch_bounds__(256) void gemm_kernel(
    const _Float16* __restrict__ GT,   // (T_, K_)
    const _Float16* __restrict__ RT,   // (S_, K_)
    float* __restrict__ out)           // (S_, T_)
{
    const int wave   = (blockIdx.x * 256 + (int)threadIdx.x) >> 5;
    const int lane   = threadIdx.x & 31;
    const int half   = lane >> 4;      // lane-group within the wave
    const int lanelo = lane & 15;

    const int wavesM = T_ / (16 * MT); // 32
    const int mTile  = wave % wavesM;
    const int nTile  = wave / wavesM;  // 0..35
    const int t0     = mTile * 16 * MT;
    const int s0     = nTile * 16 * NT;

    v8f acc[MT][NT];
#pragma unroll
    for (int mt = 0; mt < MT; ++mt)
#pragma unroll
        for (int nt = 0; nt < NT; ++nt) acc[mt][nt] = (v8f){0.f,0.f,0.f,0.f,0.f,0.f,0.f,0.f};

    // A fragment: lane(m=lanelo, half) holds K = half*16 + {0..15} contiguous
    const _Float16* aBase[MT];
#pragma unroll
    for (int mt = 0; mt < MT; ++mt)
        aBase[mt] = GT + (size_t)(t0 + mt * 16 + lanelo) * K_ + half * 16;

    // B fragment: lane(n=lanelo, half) holds K = half*16 + {0..15} contiguous
    const _Float16* bBase[NT];
#pragma unroll
    for (int nt = 0; nt < NT; ++nt)
        bBase[nt] = RT + (size_t)(s0 + nt * 16 + lanelo) * K_ + half * 16;

    for (int k0 = 0; k0 < K_; k0 += 32) {
        v16h a[MT];
#pragma unroll
        for (int mt = 0; mt < MT; ++mt) {
            v8h alo = *(const v8h*)(aBase[mt] + k0);        // 16B load
            v8h ahi = *(const v8h*)(aBase[mt] + k0 + 8);    // 16B load
            a[mt] = __builtin_shufflevector(alo, ahi,
                     0, 1, 2, 3, 4, 5, 6, 7, 8, 9, 10, 11, 12, 13, 14, 15);
        }
        v16h b[NT];
#pragma unroll
        for (int nt = 0; nt < NT; ++nt)
            b[nt] = *(const v16h*)(bBase[nt] + k0);         // 32B load

#pragma unroll
        for (int mt = 0; mt < MT; ++mt)
#pragma unroll
            for (int nt = 0; nt < NT; ++nt)
                acc[mt][nt] = __builtin_amdgcn_wmma_f32_16x16x32_f16(
                    /*neg_a=*/false, a[mt], /*neg_b=*/false, b[nt],
                    /*c_mod=*/(short)0, acc[mt][nt],
                    /*reuse_a=*/false, /*reuse_b=*/false);
    }

    // D layout: VGPR r, lanes 0-15 -> M=r, lanes 16-31 -> M=8+r; N=lanelo.
#pragma unroll
    for (int mt = 0; mt < MT; ++mt)
#pragma unroll
        for (int nt = 0; nt < NT; ++nt) {
            const int srow = s0 + nt * 16 + lanelo;
#pragma unroll
            for (int r = 0; r < 8; ++r) {
                const int trow = t0 + mt * 16 + 8 * half + r;
                out[(size_t)srow * T_ + trow] = acc[mt][nt][r];
            }
        }
}

// ---------------------------------------------------------------------------
extern "C" void kernel_launch(void* const* d_in, const int* in_sizes, int n_in,
                              void* d_out, int out_size, void* d_ws, size_t ws_size,
                              hipStream_t stream)
{
    const float* el_photons = (const float*)d_in[0];
    const float* xy         = (const float*)d_in[1];
    const float* z          = (const float*)d_in[2];
    const float* sensors    = (const float*)d_in[3];
    const float* W1         = (const float*)d_in[4];
    const float* b1         = (const float*)d_in[5];
    const float* W2         = (const float*)d_in[6];
    const float* b2         = (const float*)d_in[7];
    const float* W3         = (const float*)d_in[8];
    const float* b3         = (const float*)d_in[9];
    const float* amplitude  = (const float*)d_in[10];

    _Float16* RT = (_Float16*)d_ws;                 // S_ * K_ f16 = 18.9 MB
    _Float16* GT = RT + (size_t)S_ * K_;            // T_ * K_ f16 =  8.4 MB

    psf_kernel<<<(S_ * K_) / 256, 256, 0, stream>>>(
        el_photons, xy, sensors, W1, b1, W2, b2, W3, b3, amplitude, RT);

    gauss_kernel<<<(T_ * K_) / 256, 256, 0, stream>>>(z, GT);

    // waves = (T_/32) * (S_/64) = 32*36 = 1152 -> 144 blocks of 8 waves
    gemm_kernel<<<144, 256, 0, stream>>>(GT, RT, (float*)d_out);
}
